// HybridMambaTransformer_5652176961894
// MI455X (gfx1250) — compile-verified, matrix operand order checked by reading, not run on hardware
//
#include <hip/hip_runtime.h>
#include <stdint.h>

// ---------------------------------------------------------------------------
// Hybrid Mamba/GPT-2 forward for gfx1250 (MI455X). All matmuls go through one
// WMMA (v_wmma_f32_16x16x32_f16) tiled GEMM. The B operand tile is DMA'd into
// LDS by the Tensor Data Mover (tensor_load_to_lds + s_wait_tensorcnt) with
// D#-descriptor padding for bank-conflict-free gathers and OOB zero-fill for
// edge tiles. A-side staging is float4-vectorized fp32->f16 conversion.
// ---------------------------------------------------------------------------

typedef __attribute__((ext_vector_type(16))) _Float16 v16h;
typedef __attribute__((ext_vector_type(8)))  float    v8f;
typedef unsigned int v4u  __attribute__((ext_vector_type(4)));
typedef int          v8i_t __attribute__((ext_vector_type(8)));
typedef int          v4i_t __attribute__((ext_vector_type(4)));

#if __has_builtin(__builtin_amdgcn_tensor_load_to_lds) && \
    __has_builtin(__builtin_amdgcn_s_wait_tensorcnt)
#define HMT_HAS_TDM 1
#else
#define HMT_HAS_TDM 0
#endif

constexpr int VOCAB  = 50257;
constexpr int HMODEL = 768;
constexpr int NLAYER = 12;
constexpr int LQ     = 512;
constexpr int DINN   = 1536;
constexpr int NSS    = 16;
constexpr int RRANK  = 48;
constexpr int NHEAD  = 12;
constexpr int HDIM   = HMODEL / NHEAD;   // 64
constexpr int SSMW   = RRANK + 2 * NSS;  // 80

constexpr int BM = 64, BN = 128, BK = 64;
constexpr int LDSP = BK + 8;   // A rows: 72 halves = 144B, 16B-aligned, conflict-free
constexpr int BROW = BN + 4;   // B rows: 132 dwords (matches TDM pad 128dw + 4dw)

#define DEVFN __device__ __forceinline__

DEVFN float hmt_silu(float x)     { return x / (1.0f + __expf(-x)); }
DEVFN float hmt_softplus(float x) { return (x > 20.0f) ? x : log1pf(__expf(x)); }

DEVFN float hmt_block_sum(float v) {
  __shared__ float sm[256];
  int t = threadIdx.x;
  sm[t] = v; __syncthreads();
  for (int s = 128; s > 0; s >>= 1) { if (t < s) sm[t] += sm[t + s]; __syncthreads(); }
  float r = sm[0]; __syncthreads();
  return r;
}

DEVFN float hmt_block_max(float v) {
  __shared__ float sm[256];
  int t = threadIdx.x;
  sm[t] = v; __syncthreads();
  for (int s = 128; s > 0; s >>= 1) { if (t < s) sm[t] = fmaxf(sm[t], sm[t + s]); __syncthreads(); }
  float r = sm[0]; __syncthreads();
  return r;
}

// ---------------------------------------------------------------------------
// WMMA GEMM:  C[M,N] = A[M,K] * B  (+ bias[n])
//   bT == 0 : B stored [K,N] row-major (ldb) -> staged by TDM (tensor_load)
//   bT == 1 : B stored [N,K] row-major (computes A * B^T) -> manual staging
// Batched via blockIdx.z with element strides sA/sB/sC.
// 256 threads = 8 wave32 waves (2x4); each wave owns a 32x32 C tile via 2x2
// fragments, 2 K-substeps (BK=64) -> 8 v_wmma per iteration.
// ---------------------------------------------------------------------------
__global__ __launch_bounds__(256) void hmt_gemm_wmma(
    const float* __restrict__ A, const float* __restrict__ B,
    const float* __restrict__ bias, float* __restrict__ C,
    int M, int N, int Kd, int lda, int ldb, int ldc, int bT,
    long long sA, long long sB, long long sC)
{
  __shared__ _Float16 As[BM][LDSP];
  __shared__ float    Bs32[BK * BROW];   // [k][n] fp32, row stride BROW dwords

  const int tid  = threadIdx.x;
  const int row0 = blockIdx.y * BM;
  const int col0 = blockIdx.x * BN;
  const float* Ab = A + (long long)blockIdx.z * sA;
  const float* Bb = B + (long long)blockIdx.z * sB;
  float*       Cb = C + (long long)blockIdx.z * sC;

  const int wave = tid >> 5, lane = tid & 31;
  const int wm = wave >> 2, wn = wave & 3;     // 2 x 4 wave grid
  const int half = lane >> 4, lr = lane & 15;  // wave32 half / row-in-half

  const bool fullA = (row0 + BM <= M);
  const bool fullB = (col0 + BN <= N);

  v8f acc[2][2] = {};

  for (int k0 = 0; k0 < Kd; k0 += BK) {
    const bool fullK = (k0 + BK <= Kd);

    // ---- stage A tile: 64x64 fp32 -> f16, 16 consecutive K per thread ----
    {
      int m  = tid >> 2;
      int ks = (tid & 3) * 16;
      int gm = row0 + m;
      if (fullA && fullK) {
        const float* src = Ab + (long long)gm * lda + k0 + ks;
        if (k0 + 2 * BK <= Kd) __builtin_prefetch(src + BK, 0, 1);
#pragma unroll
        for (int jv = 0; jv < 4; ++jv) {
          float4 f = *reinterpret_cast<const float4*>(src + jv * 4);
          _Float16* dst = &As[m][ks + jv * 4];
          dst[0] = (_Float16)f.x; dst[1] = (_Float16)f.y;
          dst[2] = (_Float16)f.z; dst[3] = (_Float16)f.w;
        }
      } else {
        int cm = gm < M ? gm : (M - 1);
        const float* src = Ab + (long long)cm * lda;
#pragma unroll
        for (int j = 0; j < 16; ++j) {
          int gk = k0 + ks + j;
          int ck = gk < Kd ? gk : (Kd - 1);
          float v = src[ck];
          As[m][ks + j] = (_Float16)((gm < M && gk < Kd) ? v : 0.0f);
        }
      }
    }

    // ---- stage B tile into Bs32[k][n] ----
    if (!bT) {
#if HMT_HAS_TDM
      // Tensor Data Mover: one 2D-tile descriptor per K-step, issued by wave 0.
      // tensor_dim* give OOB zero-fill for edge tiles; pad fields insert 4
      // dwords every 128 dwords -> LDS row stride BROW = 132 dwords.
      if (wave == 0) {
        unsigned long long ga =
            (unsigned long long)(uintptr_t)(Bb + (long long)k0 * ldb + col0);
        unsigned int ldsoff = (unsigned int)(uintptr_t)&Bs32[0];
        unsigned int td0 = (unsigned int)(N - col0);   // cols remaining
        unsigned int td1 = (unsigned int)(Kd - k0);    // rows remaining
        unsigned long long st0 = (unsigned long long)ldb;
        v4u g0 = { 1u,                                  // count=1 valid descriptor
                   ldsoff,                              // lds_addr
                   (unsigned int)(ga & 0xffffffffu),
                   (unsigned int)((ga >> 32) & 0x01ffffffu) | 0x80000000u }; // type=2
        int d0 = (2 << 16)        // data_size = 4B
               | (1 << 20)        // pad_enable
               | (6 << 22)        // pad_interval: 128 dwords
               | (3 << 25);       // pad_amount:   4 dwords
        int d1 = (int)((td0 & 0xffffu) << 16);
        int d2 = (int)((td0 >> 16) & 0xffffu) | (int)((td1 & 0xffffu) << 16);
        int d3 = (int)((td1 >> 16) & 0xffffu) | (BN << 16);   // tile_dim0 = 128
        int d4 = BK;                                          // tile_dim1 = 64
        int d5 = (int)(st0 & 0xffffffffu);                    // dim0 stride lo
        int d6 = (int)((st0 >> 32) & 0xffffu);                // dim0 stride hi
        v8i_t g1 = { d0, d1, d2, d3, d4, d5, d6, 0 };
        v4i_t z4 = { 0, 0, 0, 0 };
        v8i_t z8 = { 0, 0, 0, 0, 0, 0, 0, 0 };
        __builtin_amdgcn_tensor_load_to_lds(g0, g1, z4, z4, z8, 0);
        __builtin_amdgcn_s_wait_tensorcnt(0);
      }
#else
      int k  = tid & 63;
      int nb = (tid >> 6) * 32;
      int gk = k0 + k;
      if (fullB && fullK) {
        const float* src = Bb + (long long)gk * ldb + col0 + nb;
        if (k0 + 2 * BK <= Kd) __builtin_prefetch(src + (long long)BK * ldb, 0, 1);
#pragma unroll
        for (int jv = 0; jv < 8; ++jv) {
          float4 f = *reinterpret_cast<const float4*>(src + jv * 4);
          float* dst = &Bs32[k * BROW + nb + jv * 4];
          dst[0] = f.x; dst[1] = f.y; dst[2] = f.z; dst[3] = f.w;
        }
      } else {
        int ck = gk < Kd ? gk : (Kd - 1);
        const float* src = Bb + (long long)ck * ldb;
#pragma unroll
        for (int j = 0; j < 32; ++j) {
          int gn = col0 + nb + j;
          int cn = gn < N ? gn : (N - 1);
          float v = src[cn];
          Bs32[k * BROW + nb + j] = (gn < N && gk < Kd) ? v : 0.0f;
        }
      }
#endif
    } else {
      // B^T mode (QK^T): read B[n][k0+ks..], scatter-transpose into Bs32[k][n].
      int n  = tid >> 1;
      int ks = (tid & 1) * 32;
      int gn = col0 + n;
      if (fullB && fullK) {
        const float* src = Bb + (long long)gn * ldb + k0 + ks;
        if (k0 + 2 * BK <= Kd) __builtin_prefetch(src + BK, 0, 1);
#pragma unroll
        for (int jv = 0; jv < 8; ++jv) {
          float4 f = *reinterpret_cast<const float4*>(src + jv * 4);
          Bs32[(ks + jv * 4 + 0) * BROW + n] = f.x;
          Bs32[(ks + jv * 4 + 1) * BROW + n] = f.y;
          Bs32[(ks + jv * 4 + 2) * BROW + n] = f.z;
          Bs32[(ks + jv * 4 + 3) * BROW + n] = f.w;
        }
      } else {
        int cn = gn < N ? gn : (N - 1);
        const float* src = Bb + (long long)cn * ldb;
#pragma unroll
        for (int j = 0; j < 32; ++j) {
          int gk = k0 + ks + j;
          int ck = gk < Kd ? gk : (Kd - 1);
          float v = src[ck];
          Bs32[(ks + j) * BROW + n] = (gn < N && gk < Kd) ? v : 0.0f;
        }
      }
    }
    __syncthreads();

    // ---- 2 K-substeps of 32: gather fragments (ISA 7.12.2 layout) + wmma ----
#pragma unroll
    for (int kk = 0; kk < BK; kk += 32) {
      v16h afrag[2], bfrag[2];
#pragma unroll
      for (int t = 0; t < 2; ++t) {
        const _Float16* ar = &As[wm * 32 + t * 16 + lr][kk];
        const int col = wn * 32 + t * 16 + lr;
#pragma unroll
        for (int vv = 0; vv < 8; ++vv) {
          int kb = ((vv < 4) ? vv * 2 : 16 + (vv - 4) * 2) + half * 8;
          afrag[t][2 * vv]     = ar[kb];
          afrag[t][2 * vv + 1] = ar[kb + 1];
          bfrag[t][2 * vv]     = (_Float16)Bs32[(kk + kb) * BROW + col];
          bfrag[t][2 * vv + 1] = (_Float16)Bs32[(kk + kb + 1) * BROW + col];
        }
      }
#pragma unroll
      for (int tm = 0; tm < 2; ++tm)
#pragma unroll
        for (int tn = 0; tn < 2; ++tn)
          acc[tm][tn] = __builtin_amdgcn_wmma_f32_16x16x32_f16(
              false, afrag[tm], false, bfrag[tn], (short)0, acc[tm][tn], false, false);
    }
    __syncthreads();
  }

  // ---- store: C VGPR r -> M = r + 8*half, N = lane&15 ----
  if (fullA && fullB) {
#pragma unroll
    for (int tm = 0; tm < 2; ++tm)
#pragma unroll
      for (int tn = 0; tn < 2; ++tn) {
        int gn = col0 + wn * 32 + tn * 16 + lr;
        float bv = bias ? bias[gn] : 0.0f;
#pragma unroll
        for (int r = 0; r < 8; ++r) {
          int gm = row0 + wm * 32 + tm * 16 + r + half * 8;
          Cb[(long long)gm * ldc + gn] = acc[tm][tn][r] + bv;
        }
      }
  } else {
#pragma unroll
    for (int tm = 0; tm < 2; ++tm)
#pragma unroll
      for (int tn = 0; tn < 2; ++tn) {
        int gn = col0 + wn * 32 + tn * 16 + lr;
        float bv = (bias && gn < N) ? bias[gn] : 0.0f;
#pragma unroll
        for (int r = 0; r < 8; ++r) {
          int gm = row0 + wm * 32 + tm * 16 + r + half * 8;
          if (gm < M && gn < N) Cb[(long long)gm * ldc + gn] = acc[tm][tn][r] + bv;
        }
      }
  }
}

// ---------------------------------------------------------------------------
// Elementwise / reduction kernels
// ---------------------------------------------------------------------------
__global__ __launch_bounds__(256) void hmt_embed(
    const int* __restrict__ ids, const float* __restrict__ emb,
    const float* __restrict__ pos, float* __restrict__ x)
{
  int i = blockIdx.x * 256 + threadIdx.x;
  if (i < LQ * HMODEL) {
    int l = i / HMODEL, c = i % HMODEL;
    x[i] = emb[(long long)ids[l] * HMODEL + c] + pos[i];
  }
}

__global__ __launch_bounds__(256) void hmt_layernorm(
    const float* __restrict__ x, const float* __restrict__ w,
    const float* __restrict__ b, float* __restrict__ o)
{
  int row = blockIdx.x;
  const float* xr = x + (long long)row * HMODEL;
  float* orow = o + (long long)row * HMODEL;
  float s = 0.0f;
  for (int i = threadIdx.x; i < HMODEL; i += 256) s += xr[i];
  float mean = hmt_block_sum(s) / HMODEL;
  float v = 0.0f;
  for (int i = threadIdx.x; i < HMODEL; i += 256) { float d = xr[i] - mean; v += d * d; }
  float inv = rsqrtf(hmt_block_sum(v) / HMODEL + 1e-5f);
  for (int i = threadIdx.x; i < HMODEL; i += 256)
    orow[i] = (xr[i] - mean) * inv * w[i] + b[i];
}

__global__ __launch_bounds__(256) void hmt_rmsnorm(
    const float* __restrict__ x, const float* __restrict__ w, float* __restrict__ o)
{
  int row = blockIdx.x;
  const float* xr = x + (long long)row * HMODEL;
  float* orow = o + (long long)row * HMODEL;
  float ss = 0.0f;
  for (int i = threadIdx.x; i < HMODEL; i += 256) ss += xr[i] * xr[i];
  float inv = rsqrtf(hmt_block_sum(ss) / HMODEL + 1e-5f);
  for (int i = threadIdx.x; i < HMODEL; i += 256) orow[i] = xr[i] * inv * w[i];
}

__global__ __launch_bounds__(256) void hmt_conv_silu(
    const float* __restrict__ proj, const float* __restrict__ cw,
    const float* __restrict__ cb, float* __restrict__ u)
{
  int i = blockIdx.x * 256 + threadIdx.x;
  if (i >= LQ * DINN) return;
  int l = i / DINN, d = i % DINN;
  float acc = cb[d];
#pragma unroll
  for (int j = 0; j < 4; ++j) {
    int ls = l - 3 + j;  // causal left pad K-1
    if (ls >= 0) acc += cw[d * 4 + j] * proj[(long long)ls * (2 * DINN) + d];
  }
  u[i] = hmt_silu(acc);
}

__global__ __launch_bounds__(256) void hmt_softplus_k(float* __restrict__ t, int n)
{
  int i = blockIdx.x * 256 + threadIdx.x;
  if (i < n) t[i] = hmt_softplus(t[i]);
}

// Selective scan: one thread per channel d, sequential over L, NS states in regs.
__global__ __launch_bounds__(256) void hmt_scan(
    const float* __restrict__ dt, const float* __restrict__ u,
    const float* __restrict__ ssm, const float* __restrict__ A_log,
    float* __restrict__ y)
{
  int d = blockIdx.x * 256 + threadIdx.x;
  if (d >= DINN) return;
  float Ad[NSS], h[NSS];
#pragma unroll
  for (int n = 0; n < NSS; ++n) { Ad[n] = -__expf(A_log[d * NSS + n]); h[n] = 0.0f; }
  for (int l = 0; l < LQ; ++l) {
    float dtv = dt[(long long)l * DINN + d];
    float uv  = u[(long long)l * DINN + d];
    float du  = dtv * uv;
    const float* row = ssm + (long long)l * SSMW;
    float acc = 0.0f;
#pragma unroll
    for (int n = 0; n < NSS; ++n) {
      float dA = __expf(dtv * Ad[n]);
      h[n] = dA * h[n] + du * row[RRANK + n];       // B part
      acc += h[n] * row[RRANK + NSS + n];           // C part
    }
    y[(long long)l * DINN + d] = acc;
  }
}

__global__ __launch_bounds__(256) void hmt_gate(
    float* __restrict__ y, const float* __restrict__ u,
    const float* __restrict__ Dp, const float* __restrict__ proj)
{
  int i = blockIdx.x * 256 + threadIdx.x;
  if (i >= LQ * DINN) return;
  int l = i / DINN, d = i % DINN;
  float g = proj[(long long)l * (2 * DINN) + DINN + d];
  y[i] = (y[i] + u[i] * Dp[d]) * hmt_silu(g);
}

__global__ __launch_bounds__(256) void hmt_resadd(
    float* __restrict__ x, const float* __restrict__ src, const float* __restrict__ sc)
{
  int i = blockIdx.x * 256 + threadIdx.x;
  if (i < LQ * HMODEL) x[i] += sc[0] * src[i];
}

__global__ __launch_bounds__(256) void hmt_split_heads(
    const float* __restrict__ qkv, float* __restrict__ Qh,
    float* __restrict__ Kh, float* __restrict__ Vh)
{
  int i = blockIdx.x * 256 + threadIdx.x;
  if (i >= LQ * HMODEL) return;
  int l = i / HMODEL, c = i % HMODEL;
  int h = c / HDIM, dd = c % HDIM;
  long long o = ((long long)h * LQ + l) * HDIM + dd;
  long long base = (long long)l * 3 * HMODEL;
  Qh[o] = qkv[base + c];
  Kh[o] = qkv[base + HMODEL + c];
  Vh[o] = qkv[base + 2 * HMODEL + c];
}

__global__ __launch_bounds__(256) void hmt_softmax_causal(float* __restrict__ S)
{
  int h = blockIdx.x, q = blockIdx.y;
  float* row = S + ((long long)h * LQ + q) * LQ;
  const float scale = 0.125f;  // 1/sqrt(64)
  float mx = -3.4e38f;
  for (int k = threadIdx.x; k <= q; k += 256) mx = fmaxf(mx, row[k] * scale);
  mx = hmt_block_max(mx);
  float sum = 0.0f;
  for (int k = threadIdx.x; k <= q; k += 256) sum += __expf(row[k] * scale - mx);
  sum = hmt_block_sum(sum);
  float inv = 1.0f / sum;
  for (int k = threadIdx.x; k < LQ; k += 256)
    row[k] = (k <= q) ? __expf(row[k] * scale - mx) * inv : 0.0f;
}

__global__ __launch_bounds__(256) void hmt_merge_heads(
    const float* __restrict__ Oh, float* __restrict__ out)
{
  int i = blockIdx.x * 256 + threadIdx.x;
  if (i >= LQ * HMODEL) return;
  int l = i / HMODEL, c = i % HMODEL;
  int h = c / HDIM, dd = c % HDIM;
  out[i] = Oh[((long long)h * LQ + l) * HDIM + dd];
}

__global__ __launch_bounds__(256) void hmt_nll(
    const float* __restrict__ logits, const int* __restrict__ labels,
    float* __restrict__ nll)
{
  int l = blockIdx.x;
  const float* r = logits + (long long)l * VOCAB;
  float mx = -3.4e38f;
  for (int i = threadIdx.x; i < VOCAB; i += 256) mx = fmaxf(mx, r[i]);
  mx = hmt_block_max(mx);
  float s = 0.0f;
  for (int i = threadIdx.x; i < VOCAB; i += 256) s += __expf(r[i] - mx);
  s = hmt_block_sum(s);
  if (threadIdx.x == 0) nll[l] = -(r[labels[l]] - mx - logf(s));
}

__global__ __launch_bounds__(256) void hmt_loss(
    const float* __restrict__ nll, float* __restrict__ out)
{
  float v = nll[threadIdx.x] + nll[threadIdx.x + 256];
  v = hmt_block_sum(v);
  if (threadIdx.x == 0) out[0] = v / (float)LQ;
}

// ---------------------------------------------------------------------------
// Host orchestration
// ---------------------------------------------------------------------------
namespace {
struct MP {
  const float *norm_w, *in_proj_w, *conv_w, *conv_b, *x_proj_w,
              *dt_proj_w, *dt_proj_b, *A_log, *D_param, *out_proj_w;
};
}

extern "C" void kernel_launch(void* const* d_in, const int* in_sizes, int n_in,
                              void* d_out, int out_size, void* d_ws, size_t ws_size,
                              hipStream_t stream) {
  (void)in_sizes; (void)n_in; (void)out_size; (void)ws_size;

  // Input flattening assumption: depth-first dict insertion order.
  //  0 embed, 1 pos_embed, 2..11 initial_mamba (norm_w,in_proj_w,conv_w,conv_b,
  //  x_proj_w,dt_proj_w,dt_proj_b,A_log,D_param,out_proj_w), 12 init_gate,
  //  13+20*i per layer: ln1_w,ln1_b,ln2_w,ln2_b,c_attn_w,c_attn_b,c_proj_w,
  //  c_proj_b, mamba(10), gate_a, gate_b; 253 ln_f_w, 254 ln_f_b,
  //  255 lm_head_w, 256 input_ids, 257 labels.
  auto F = [&](int i) -> const float* { return (const float*)d_in[i]; };
  const int* input_ids = (const int*)d_in[256];
  const int* labels    = (const int*)d_in[257];

  float* w = (float*)d_ws;
  auto take = [&](long long n) { float* p = w; w += n; return p; };
  float* x    = take((long long)LQ * HMODEL);
  float* lno  = take((long long)LQ * HMODEL);
  float* nrm  = take((long long)LQ * HMODEL);
  float* mo   = take((long long)LQ * HMODEL);
  float* qkv  = take((long long)LQ * 3 * HMODEL);
  float* Qh   = take((long long)LQ * HMODEL);
  float* Kh   = take((long long)LQ * HMODEL);
  float* Vh   = take((long long)LQ * HMODEL);
  float* Sb   = take((long long)NHEAD * LQ * LQ);
  float* Oh   = take((long long)LQ * HMODEL);
  float* mrg  = take((long long)LQ * HMODEL);
  float* proj = take((long long)LQ * 2 * DINN);
  float* u    = take((long long)LQ * DINN);
  float* ssm  = take((long long)LQ * SSMW);
  float* dtb  = take((long long)LQ * DINN);
  float* yb   = take((long long)LQ * DINN);
  float* nllb = take((long long)LQ);

  auto gemm = [&](const float* A, const float* B, const float* bias, float* C,
                  int M, int N, int Kd, int lda, int ldb, int ldc, int bT,
                  int batch, long long sA, long long sB, long long sC) {
    dim3 g((N + BN - 1) / BN, (M + BM - 1) / BM, batch);
    hmt_gemm_wmma<<<g, 256, 0, stream>>>(A, B, bias, C, M, N, Kd, lda, ldb, ldc,
                                         bT, sA, sB, sC);
  };
  const int EW = (LQ * HMODEL + 255) / 256;
  const int ED = (LQ * DINN + 255) / 256;

  auto mp = [&](int b) -> MP {
    return MP{F(b), F(b + 1), F(b + 2), F(b + 3), F(b + 4),
              F(b + 5), F(b + 6), F(b + 7), F(b + 8), F(b + 9)};
  };

  auto run_mamba = [&](const float* xin, float* outb, const MP& p) {
    hmt_rmsnorm<<<LQ, 256, 0, stream>>>(xin, p.norm_w, nrm);
    gemm(nrm, p.in_proj_w, nullptr, proj, LQ, 2 * DINN, HMODEL,
         HMODEL, 2 * DINN, 2 * DINN, 0, 1, 0, 0, 0);
    hmt_conv_silu<<<ED, 256, 0, stream>>>(proj, p.conv_w, p.conv_b, u);
    gemm(u, p.x_proj_w, nullptr, ssm, LQ, SSMW, DINN, DINN, SSMW, SSMW, 0, 1, 0, 0, 0);
    gemm(ssm, p.dt_proj_w, p.dt_proj_b, dtb, LQ, DINN, RRANK,
         SSMW, DINN, DINN, 0, 1, 0, 0, 0);
    hmt_softplus_k<<<ED, 256, 0, stream>>>(dtb, LQ * DINN);
    hmt_scan<<<(DINN + 255) / 256, 256, 0, stream>>>(dtb, u, ssm, p.A_log, yb);
    hmt_gate<<<ED, 256, 0, stream>>>(yb, u, p.D_param, proj);
    gemm(yb, p.out_proj_w, nullptr, outb, LQ, HMODEL, DINN,
         DINN, HMODEL, HMODEL, 0, 1, 0, 0, 0);
  };

  auto run_attn = [&](const float* xin, float* outb, const float* caw,
                      const float* cab, const float* cpw, const float* cpb) {
    gemm(xin, caw, cab, qkv, LQ, 3 * HMODEL, HMODEL,
         HMODEL, 3 * HMODEL, 3 * HMODEL, 0, 1, 0, 0, 0);
    hmt_split_heads<<<EW, 256, 0, stream>>>(qkv, Qh, Kh, Vh);
    gemm(Qh, Kh, nullptr, Sb, LQ, LQ, HDIM, HDIM, HDIM, LQ, /*bT=*/1, NHEAD,
         (long long)LQ * HDIM, (long long)LQ * HDIM, (long long)LQ * LQ);
    hmt_softmax_causal<<<dim3(NHEAD, LQ), 256, 0, stream>>>(Sb);
    gemm(Sb, Vh, nullptr, Oh, LQ, HDIM, LQ, LQ, HDIM, HDIM, 0, NHEAD,
         (long long)LQ * LQ, (long long)LQ * HDIM, (long long)LQ * HDIM);
    hmt_merge_heads<<<EW, 256, 0, stream>>>(Oh, mrg);
    gemm(mrg, cpw, cpb, outb, LQ, HMODEL, HMODEL,
         HMODEL, HMODEL, HMODEL, 0, 1, 0, 0, 0);
  };

  // Embedding + initial mamba
  hmt_embed<<<EW, 256, 0, stream>>>(input_ids, F(0), F(1), x);
  MP im = mp(2);
  run_mamba(x, mo, im);
  hmt_resadd<<<EW, 256, 0, stream>>>(x, mo, F(12));

  for (int i = 0; i < NLAYER; ++i) {
    int b = 13 + 20 * i;
    const float *ln1w = F(b + 0), *ln1b = F(b + 1);
    const float *ln2w = F(b + 2), *ln2b = F(b + 3);
    const float *caw = F(b + 4), *cab = F(b + 5);
    const float *cpw = F(b + 6), *cpb = F(b + 7);
    MP lm = mp(b + 8);
    const float *ga = F(b + 18), *gb = F(b + 19);
    if (i < NLAYER / 2) {  // mamba first
      hmt_layernorm<<<LQ, 256, 0, stream>>>(x, ln1w, ln1b, lno);
      run_mamba(lno, mo, lm);
      hmt_resadd<<<EW, 256, 0, stream>>>(x, mo, ga);
      hmt_layernorm<<<LQ, 256, 0, stream>>>(x, ln2w, ln2b, lno);
      run_attn(lno, mo, caw, cab, cpw, cpb);
      hmt_resadd<<<EW, 256, 0, stream>>>(x, mo, gb);
    } else {               // attention first
      hmt_layernorm<<<LQ, 256, 0, stream>>>(x, ln1w, ln1b, lno);
      run_attn(lno, mo, caw, cab, cpw, cpb);
      hmt_resadd<<<EW, 256, 0, stream>>>(x, mo, ga);
      hmt_layernorm<<<LQ, 256, 0, stream>>>(x, ln2w, ln2b, lno);
      run_mamba(lno, mo, lm);
      hmt_resadd<<<EW, 256, 0, stream>>>(x, mo, gb);
    }
  }

  // Final LN + lm_head + loss. Logits live at d_out+1, mean NLL at d_out[0].
  hmt_layernorm<<<LQ, 256, 0, stream>>>(x, F(253), F(254), lno);
  float* logits = (float*)d_out + 1;
  gemm(lno, F(255), nullptr, logits, LQ, VOCAB, HMODEL,
       HMODEL, VOCAB, VOCAB, 0, 1, 0, 0, 0);
  hmt_nll<<<LQ, 256, 0, stream>>>(logits, labels, nllb);
  hmt_loss<<<1, 256, 0, stream>>>(nllb, (float*)d_out);
}
